// TransConvLayer_64235530879069
// MI455X (gfx1250) — matrix operand order repaired and checked by reading, not compiled, not run on hardware
//
#include <hip/hip_runtime.h>
#include <hip/hip_bf16.h>

// Problem constants
#define NS    50000
#define NP    50016           // NS padded to multiple of 32 for K-loops
#define DIN0  256
#define HH    4
#define DH    256
#define HD    1024            // HH*DH
#define QKV   3072
#define NSPLIT 16

typedef __attribute__((ext_vector_type(16))) __bf16 v16bf;
typedef __attribute__((ext_vector_type(8)))  float  v8f;

union Frag  { uint4 q[2]; v16bf bf; };
union Pack8 { unsigned short u[8]; uint4 v; };

__device__ __forceinline__ unsigned short f2bf(float f) {
  unsigned int u = __float_as_uint(f);
  u += 0x7FFFu + ((u >> 16) & 1u);            // round-to-nearest-even
  return (unsigned short)(u >> 16);
}
__device__ __forceinline__ float bf2f(unsigned short h) {
  return __uint_as_float(((unsigned int)h) << 16);
}

// A fragment: 16x32 bf16, row-major source, lane l: row = l%16,
// K runs [k0 + (l/16)*8, +8) and [k0 + 16 + (l/16)*8, +8)  (ISA 7.12.2)
__device__ __forceinline__ v16bf load_frag_a(const unsigned short* base, int ld, int row, int k0) {
  int l = threadIdx.x & 31;
  const unsigned short* p = base + (size_t)(row + (l & 15)) * ld + k0 + ((l >> 4) << 3);
  Frag f;
  f.q[0] = *(const uint4*)p;
  f.q[1] = *(const uint4*)(p + 16);
  return f.bf;
}

// B fragment: 32x16 bf16 from "Bt" storage (output-column major, contiguous K):
// lane l: column = col0 + l%16, K runs [k0 + (l/16)*8, +8) and [k0 + 16 + (l/16)*8, +8)
__device__ __forceinline__ v16bf load_frag_b(const unsigned short* bt, int ld, int col0, int k0) {
  int l = threadIdx.x & 31;
  const unsigned short* p = bt + (size_t)(col0 + (l & 15)) * ld + k0 + ((l >> 4) << 3);
  Frag f;
  f.q[0] = *(const uint4*)p;
  f.q[1] = *(const uint4*)(p + 16);
  return f.bf;
}

__device__ __forceinline__ v8f wmma_bf16(v16bf a, v16bf b, v8f c) {
  return __builtin_amdgcn_wmma_f32_16x16x32_bf16(false, a, false, b, (short)0, c, false, false);
}

// ---------------- K0: convert inputs to bf16, zero accumulators & pads ----
__global__ void k0_init(const float* __restrict__ x,
                        const float* __restrict__ Wq, const float* __restrict__ Wqb,
                        const float* __restrict__ Wk, const float* __restrict__ Wkb,
                        const float* __restrict__ Wv, const float* __restrict__ Wvb,
                        unsigned short* __restrict__ xb, unsigned short* __restrict__ Wb,
                        float* __restrict__ biasc,
                        unsigned short* __restrict__ kT, unsigned short* __restrict__ vT,
                        float* __restrict__ kvs_raw, float* __restrict__ ksum,
                        float* __restrict__ scal) {
  size_t i = (size_t)blockIdx.x * blockDim.x + threadIdx.x;
  if (i < (size_t)NS * DIN0) xb[i] = f2bf(x[i]);
  if (i < (size_t)QKV * DIN0) {
    int j = (int)(i >> 8), k = (int)(i & 255);
    float w = (j < HD)   ? Wq[(size_t)j * DIN0 + k]
            : (j < 2*HD) ? Wk[(size_t)(j - HD) * DIN0 + k]
                         : Wv[(size_t)(j - 2*HD) * DIN0 + k];
    Wb[i] = f2bf(w);
  }
  if (i < QKV)
    biasc[i] = (i < HD) ? Wqb[i] : (i < 2*HD) ? Wkb[i - HD] : Wvb[i - 2*HD];
  if (i < (size_t)HH * DH * DH) kvs_raw[i] = 0.f;
  if (i < HD) ksum[i] = 0.f;
  if (i < 4) scal[i] = 0.f;
  if (i < (size_t)HD * (NP - NS)) {           // zero K-padding tails
    int f = (int)(i >> 4), off = (int)(i & 15);
    kT[(size_t)f * NP + NS + off] = 0;
    vT[(size_t)f * NP + NS + off] = 0;
  }
}

// ---------------- K1: QKV projection GEMM (WMMA) --------------------------
// grid (3125, 12), block 128 (4 waves). Wave: 16 samples x 64 cols.
__global__ __launch_bounds__(128) void k1_qkv(
    const unsigned short* __restrict__ xb, const unsigned short* __restrict__ Wb,
    const float* __restrict__ biasc,
    unsigned short* __restrict__ qo, unsigned short* __restrict__ kT,
    unsigned short* __restrict__ vT, unsigned short* __restrict__ vb,
    float* __restrict__ scal) {
  int w = threadIdx.x >> 5;
  int sbase = blockIdx.x * 16;
  int cbase = blockIdx.y * 256 + w * 64;
  v8f acc[4] = {};
  for (int k0 = 0; k0 < DIN0; k0 += 32) {
    v16bf a = load_frag_a(xb, DIN0, sbase, k0);
#pragma unroll
    for (int t = 0; t < 4; ++t) {
      v16bf b = load_frag_b(Wb, DIN0, cbase + 16 * t, k0);
      acc[t] = wmma_bf16(a, b, acc[t]);
    }
  }
  int l = threadIdx.x & 31;
  int n = l & 15, half = l >> 4;
  float ss = 0.f;
  if (cbase < HD) {                                    // ---- Q: row-major bf16 + ||q||^2
#pragma unroll
    for (int t = 0; t < 4; ++t) {
      int col = cbase + 16 * t + n;
      float bias = biasc[col];
#pragma unroll
      for (int r = 0; r < 8; ++r) {
        float fv = acc[t][r] + bias;
        ss += fv * fv;
        qo[(size_t)(sbase + 8 * half + r) * HD + col] = f2bf(fv);
      }
    }
    for (int off = 16; off > 0; off >>= 1) ss += __shfl_xor(ss, off);
    if (l == 0) atomicAdd(&scal[0], ss);
  } else if (cbase < 2 * HD) {                         // ---- K: transposed bf16 + ||k||^2
#pragma unroll
    for (int t = 0; t < 4; ++t) {
      int col = cbase + 16 * t + n;
      float bias = biasc[col];
      int f = col - HD;
      Pack8 pk;
#pragma unroll
      for (int r = 0; r < 8; ++r) {
        float fv = acc[t][r] + bias;
        ss += fv * fv;
        pk.u[r] = f2bf(fv);
      }
      *(uint4*)(kT + (size_t)f * NP + sbase + 8 * half) = pk.v;   // 16B contiguous
    }
    for (int off = 16; off > 0; off >>= 1) ss += __shfl_xor(ss, off);
    if (l == 0) atomicAdd(&scal[1], ss);
  } else {                                             // ---- V: transposed + row-major bf16
#pragma unroll
    for (int t = 0; t < 4; ++t) {
      int col = cbase + 16 * t + n;
      float bias = biasc[col];
      int f = col - 2 * HD;
      Pack8 pk;
#pragma unroll
      for (int r = 0; r < 8; ++r) {
        float fv = acc[t][r] + bias;
        pk.u[r] = f2bf(fv);
        vb[(size_t)(sbase + 8 * half + r) * HD + f] = pk.u[r];
      }
      *(uint4*)(vT + (size_t)f * NP + sbase + 8 * half) = pk.v;
    }
  }
}

// ---------------- K2: ksum[f] = sum_n k[n][f]  (reads kT rows) ------------
__global__ __launch_bounds__(256) void k2_ksum(const unsigned short* __restrict__ kT,
                                               float* __restrict__ ksum) {
  __shared__ float red[256];
  int f = blockIdx.x;
  float s = 0.f;
  for (int nn = threadIdx.x; nn < NP; nn += 256) s += bf2f(kT[(size_t)f * NP + nn]);
  red[threadIdx.x] = s;
  __syncthreads();
  for (int o = 128; o > 0; o >>= 1) {
    if (threadIdx.x < o) red[threadIdx.x] += red[threadIdx.x + o];
    __syncthreads();
  }
  if (threadIdx.x == 0) ksum[f] = red[0];
}

// ---------------- K3: kvs_raw[h] += k_h^T @ v_h  (split-K WMMA) -----------
// grid (16, 4, NSPLIT), block 128. Wave: 16 m-rows x 64 d-cols.
__global__ __launch_bounds__(128) void k3_kvs(const unsigned short* __restrict__ kT,
                                              const unsigned short* __restrict__ vT,
                                              float* __restrict__ kvs_raw) {
  int w = threadIdx.x >> 5;
  int h = blockIdx.y;
  int mbase = h * DH + blockIdx.x * 16;
  int dbase = w * 64;
  v8f acc[4] = {};
  for (int ks = blockIdx.z; ks < NP / 32; ks += NSPLIT) {
    int k0 = ks * 32;
    v16bf a = load_frag_a(kT, NP, mbase, k0);
#pragma unroll
    for (int t = 0; t < 4; ++t) {
      v16bf b = load_frag_b(vT, NP, h * DH + dbase + 16 * t, k0);
      acc[t] = wmma_bf16(a, b, acc[t]);
    }
  }
  int l = threadIdx.x & 31;
  int n = l & 15, half = l >> 4;
  int mloc = blockIdx.x * 16 + 8 * half;
#pragma unroll
  for (int t = 0; t < 4; ++t) {
    int dloc = dbase + 16 * t + n;
#pragma unroll
    for (int r = 0; r < 8; ++r)
      atomicAdd(&kvs_raw[(size_t)h * DH * DH + (size_t)(mloc + r) * DH + dloc], acc[t][r]);
  }
}

// ---------------- K4: kvsbT[h][d][m] = bf16(kvs_raw[h][m][d]) -------------
__global__ void k4_kvsb(const float* __restrict__ kvs_raw, unsigned short* __restrict__ kvsbT) {
  int i = blockIdx.x * blockDim.x + threadIdx.x;
  if (i >= HH * DH * DH) return;
  int h = i >> 16, rem = i & 65535, m = rem >> 8, d = rem & 255;
  kvsbT[(size_t)h * DH * DH + (size_t)d * DH + m] = f2bf(kvs_raw[i]);
}

// ---------------- K5: attention apply + mean + residual + LN + ReLU -------
// grid 3125, block 256 (8 waves, 16 samples/block). Wave w: head w/2, d-half (w&1)*128.
__global__ __launch_bounds__(256) void k5_final(
    const unsigned short* __restrict__ qo, const unsigned short* __restrict__ kvsbT,
    const unsigned short* __restrict__ vb, const float* __restrict__ x,
    const float* __restrict__ ksum, const float* __restrict__ scal,
    const float* __restrict__ ln_g, const float* __restrict__ ln_b,
    float* __restrict__ out) {
  __shared__ float normLds[16][4];
  __shared__ float obuf[16][DH];
  __shared__ float red[16][16];
  __shared__ float smu[16], srs[16];
  int tid = threadIdx.x;
  int sbase = blockIdx.x * 16;
  if (tid < 64) normLds[tid >> 2][tid & 3] = 0.f;
  for (int j = 0; j < 16; ++j) {                       // residual seed
    int e = j * 256 + tid, s = e >> 8, d = e & 255;
    obuf[s][d] = x[(size_t)(sbase + s) * DIN0 + d];
  }
  __syncthreads();
  {                                                    // normalizer dot q·ksum per (s,h)
    int p = tid & 63, c = tid >> 6;
    int s = p >> 2, h = p & 3;
    float part = 0.f;
    const unsigned short* qrow = qo + (size_t)(sbase + s) * HD + h * DH + c * 64;
    const float* krow = ksum + h * DH + c * 64;
    for (int j = 0; j < 64; ++j) part += bf2f(qrow[j]) * krow[j];
    atomicAdd(&normLds[s][h], part);
  }
  __syncthreads();
  int w = tid >> 5, h = w >> 1, dhalf = (w & 1) * 128;
  v8f acc[8] = {};
  for (int k0 = 0; k0 < DH; k0 += 32) {
    v16bf a = load_frag_a(qo, HD, sbase, h * DH + k0);
#pragma unroll
    for (int t = 0; t < 8; ++t) {
      v16bf b = load_frag_b(kvsbT + (size_t)h * DH * DH, DH, dhalf + 16 * t, k0);
      acc[t] = wmma_bf16(a, b, acc[t]);
    }
  }
  float rqrk = rsqrtf(scal[0]) * rsqrtf(scal[1]);
  const float nf = 50000.0f;
  int l = tid & 31, n = l & 15, half = l >> 4;
#pragma unroll
  for (int t = 0; t < 8; ++t) {
    int dl = dhalf + 16 * t + n;                       // d within head
    int col = h * DH + dl;
#pragma unroll
    for (int r = 0; r < 8; ++r) {
      int s = 8 * half + r;
      float vv  = bf2f(vb[(size_t)(sbase + s) * HD + col]);
      float num = rqrk * acc[t][r] + nf * vv;
      float den = rqrk * normLds[s][h] + nf;
      atomicAdd(&obuf[s][dl], 0.25f * num / den);      // head mean
    }
  }
  __syncthreads();
  int s = tid >> 4, l16 = tid & 15;                    // LayerNorm: 16 threads/sample
  float part = 0.f;
  for (int j = 0; j < 16; ++j) part += obuf[s][l16 * 16 + j];
  red[s][l16] = part;
  __syncthreads();
  if (tid < 16) {
    float m = 0.f;
    for (int j = 0; j < 16; ++j) m += red[tid][j];
    smu[tid] = m * (1.0f / 256.0f);
  }
  __syncthreads();
  float mu = smu[s];
  part = 0.f;
  for (int j = 0; j < 16; ++j) { float d = obuf[s][l16 * 16 + j] - mu; part += d * d; }
  red[s][l16] = part;
  __syncthreads();
  if (tid < 16) {
    float vs = 0.f;
    for (int j = 0; j < 16; ++j) vs += red[tid][j];
    srs[tid] = rsqrtf(vs * (1.0f / 256.0f) + 1e-5f);
  }
  __syncthreads();
  float rstd = srs[s];
  for (int j = 0; j < 16; ++j) {
    int d = l16 * 16 + j;
    float val = (obuf[s][d] - mu) * rstd * ln_g[d] + ln_b[d];
    out[(size_t)(sbase + s) * DIN0 + d] = fmaxf(val, 0.f);
  }
}

extern "C" void kernel_launch(void* const* d_in, const int* in_sizes, int n_in,
                              void* d_out, int out_size, void* d_ws, size_t ws_size,
                              hipStream_t stream) {
  const float* x   = (const float*)d_in[0];
  const float* Wq  = (const float*)d_in[1];
  const float* Wqb = (const float*)d_in[2];
  const float* Wk  = (const float*)d_in[3];
  const float* Wkb = (const float*)d_in[4];
  const float* Wv  = (const float*)d_in[5];
  const float* Wvb = (const float*)d_in[6];
  const float* lng = (const float*)d_in[7];
  const float* lnb = (const float*)d_in[8];
  float* out = (float*)d_out;

  char* ws = (char*)d_ws;
  size_t o = 0;
  unsigned short* xb    = (unsigned short*)(ws + o); o += (size_t)NS * DIN0 * 2;
  unsigned short* Wb    = (unsigned short*)(ws + o); o += (size_t)QKV * DIN0 * 2;
  float*          biasc = (float*)(ws + o);          o += (size_t)QKV * 4;
  unsigned short* qb    = (unsigned short*)(ws + o); o += (size_t)NS * HD * 2;
  unsigned short* kT    = (unsigned short*)(ws + o); o += (size_t)HD * NP * 2;
  unsigned short* vT    = (unsigned short*)(ws + o); o += (size_t)HD * NP * 2;
  unsigned short* vb    = (unsigned short*)(ws + o); o += (size_t)NS * HD * 2;
  float*          kvs   = (float*)(ws + o);          o += (size_t)HH * DH * DH * 4;
  unsigned short* kvsbT = (unsigned short*)(ws + o); o += (size_t)HH * DH * DH * 2;
  float*          ksum  = (float*)(ws + o);          o += (size_t)HD * 4;
  float*          scal  = (float*)(ws + o);          o += 64;
  (void)ws_size; (void)in_sizes; (void)n_in; (void)out_size;

  unsigned g0 = (unsigned)(((size_t)NS * DIN0 + 255) / 256);
  k0_init<<<dim3(g0), 256, 0, stream>>>(x, Wq, Wqb, Wk, Wkb, Wv, Wvb,
                                        xb, Wb, biasc, kT, vT, kvs, ksum, scal);
  k1_qkv <<<dim3(NS / 16, QKV / 256), 128, 0, stream>>>(xb, Wb, biasc, qb, kT, vT, vb, scal);
  k2_ksum<<<dim3(HD), 256, 0, stream>>>(kT, ksum);
  k3_kvs <<<dim3(DH / 16, HH, NSPLIT), 128, 0, stream>>>(kT, vT, kvs);
  k4_kvsb<<<dim3((HH * DH * DH + 255) / 256), 256, 0, stream>>>(kvs, kvsbT);
  k5_final<<<dim3(NS / 16), 256, 0, stream>>>(qb, kvsbT, vb, x, ksum, scal, lng, lnb, out);
}
